// SparseSerialThorMoE_7533372638013
// MI455X (gfx1250) — compile-verified, matrix operand order checked by reading, not run on hardware
//
#include <hip/hip_runtime.h>
#include <math.h>

// ---------------- problem constants ----------------
constexpr int E   = 16;
constexpr int T   = 8192;
constexpr int H   = 1024;
constexpr int I   = 4096;
constexpr int TPE = T / E;          // 512 tokens per expert

// ---------------- types ----------------
typedef __bf16 bf16_t;
typedef __attribute__((ext_vector_type(4)))  bf16_t bf16x4;
typedef __attribute__((ext_vector_type(16))) bf16_t v16bf;
typedef __attribute__((ext_vector_type(8)))  float  v8f;
typedef __attribute__((ext_vector_type(4)))  int    v4i_t;
typedef __attribute__((ext_vector_type(4)))  float  f32x4;

__device__ __forceinline__ bf16_t to_bf16(float f) { return (bf16_t)f; }

__device__ __forceinline__ float gelu_exact(float x) {
    return 0.5f * x * (1.0f + erff(x * 0.70710678118654752f));
}

// ---------------- explicit global (addrspace 1) accessors ----------------
// Keeps weight loads on the GLOBAL path (LOADcnt only). A generic pointer
// would emit flat_load_*, which also ticks DScnt and makes every
// s_wait_dscnt before the WMMAs falsely wait on in-flight weight loads.
__device__ __forceinline__ f32x4 load_g_f32x4(const float* p) {
    return *(const f32x4 __attribute__((address_space(1)))*)p;
}
__device__ __forceinline__ void store_g_bf16(bf16_t* p, bf16_t v) {
    *(bf16_t __attribute__((address_space(1)))*)p = v;
}
__device__ __forceinline__ void store_g_f32(float* p, float v) {
    *(float __attribute__((address_space(1)))*)p = v;
}

// ---------------- async global->LDS (gfx1250 GLOBAL_LOAD_ASYNC_TO_LDS_B128) ----------------
#if __has_builtin(__builtin_amdgcn_global_load_async_to_lds_b128) && \
    __has_builtin(__builtin_amdgcn_s_wait_asynccnt)
#define MOE_ASYNC_LDS 1
#endif

#ifdef MOE_ASYNC_LDS
// builtin signature (from hipcc diagnostics): (v4i as1*, v4i as3*, Imm i32 offset, Imm i32 cpol)
typedef v4i_t __attribute__((address_space(1)))* g_v4i_ptr;   // non-const!
typedef v4i_t __attribute__((address_space(3)))* l_v4i_ptr;
__device__ __forceinline__ void copy16_g2l(const bf16_t* g, bf16_t* l) {
    __builtin_amdgcn_global_load_async_to_lds_b128(
        (g_v4i_ptr)g, (l_v4i_ptr)l, /*imm offset*/0, /*cpol*/0);
}
__device__ __forceinline__ void wait_async() {
    __builtin_amdgcn_s_wait_asynccnt(0);
}
#else
__device__ __forceinline__ void copy16_g2l(const bf16_t* g, bf16_t* l) {
    *(uint4*)l = *(const uint4*)g;
}
__device__ __forceinline__ void wait_async() {}
#endif

// =====================================================================
// Kernel 0: gather tokens by perm, cast fp32 -> bf16.
// =====================================================================
__global__ __launch_bounds__(256)
void moe_gather_cast(const float* __restrict__ hs, const int* __restrict__ perm,
                     bf16_t* __restrict__ A1) {
    const int s   = blockIdx.x;
    const int src = perm[s];
    const f32x4 f = load_g_f32x4(hs + (size_t)src * H + threadIdx.x * 4);
    bf16x4 h;
    h.x = to_bf16(f.x); h.y = to_bf16(f.y); h.z = to_bf16(f.z); h.w = to_bf16(f.w);
    *(bf16x4 __attribute__((address_space(1)))*)(A1 + (size_t)s * H + threadIdx.x * 4) = h;
}

// =====================================================================
// Double-buffered bf16 WMMA GEMM core.
//   block tile 128(M) x 128(N), K-step 32, 256 threads = 8 waves,
//   wave (wm,wn) owns 64x32 -> 4x2 v_wmma_f32_16x16x32_bf16.
//   A tile (bf16): async DMA global->LDS (ASYNCcnt).
//   B tile (fp32->bf16): clustered global_load_b128 -> cvt -> ds_store,
//   issued for tile k+1 while tile k's WMMAs execute.
// =====================================================================
constexpr int LDSPAD = 40;   // bf16 row stride: 80B, keeps 16B-aligned frags

template<int K>
__device__ __forceinline__ void wmma_gemm_core(
    const bf16_t* __restrict__ A, const float* __restrict__ B,
    v8f (&acc)[4][2], int& wm_o, int& wn_o, int& lr_o, int& lane_o)
{
    __shared__ bf16_t As[2][128][LDSPAD];
    __shared__ bf16_t Bs[2][128][LDSPAD];

    const int tid  = threadIdx.x;
    const int lane = tid & 31;
    const int wave = tid >> 5;
    const int wm   = wave & 1;
    const int wn   = wave >> 1;
    const int lr   = lane & 15;
    const int kh   = (lane >> 4) << 4;
    wm_o = wm; wn_o = wn; lr_o = lr; lane_o = lane;

    // ---- A: 2 chunks of 16B (8 bf16) per thread (128x32 tile) ----
    const int ar0 = tid >> 2,          ac0 = (tid & 3) * 8;
    const int ar1 = (tid + 256) >> 2,  ac1 = ((tid + 256) & 3) * 8;
    const bf16_t* gA0 = A + (size_t)ar0 * K + ac0;
    const bf16_t* gA1 = A + (size_t)ar1 * K + ac1;

    // ---- B: 4 chunks of float4 per thread (128x32 tile) ----
    int brow[4], bcol[4];
    const float* gB[4];
    #pragma unroll
    for (int r = 0; r < 4; ++r) {
        const int c = tid + r * 256;
        brow[r] = c >> 3;
        bcol[r] = (c & 7) * 4;
        gB[r]   = B + (size_t)brow[r] * K + bcol[r];
    }

    // ---- prologue: fill buffer 0 ----
    {
        f32x4 fb[4];
        #pragma unroll
        for (int r = 0; r < 4; ++r) fb[r] = load_g_f32x4(gB[r]);
        copy16_g2l(gA0, &As[0][ar0][ac0]);
        copy16_g2l(gA1, &As[0][ar1][ac1]);
        gA0 += 32; gA1 += 32;
        #pragma unroll
        for (int r = 0; r < 4; ++r) {
            bf16x4 hv;
            hv.x = to_bf16(fb[r].x); hv.y = to_bf16(fb[r].y);
            hv.z = to_bf16(fb[r].z); hv.w = to_bf16(fb[r].w);
            *(bf16x4*)&Bs[0][brow[r]][bcol[r]] = hv;
            gB[r] += 32;
        }
        wait_async();
        __syncthreads();
    }

    constexpr int NSTEP = K / 32;
    for (int ks = 0; ks < NSTEP; ++ks) {
        const int cur = ks & 1;
        const int nxt = cur ^ 1;
        const bool has_next = (ks + 1 < NSTEP);

        // ---- issue next tile's traffic before compute ----
        f32x4 fb[4];
        if (has_next) {
            copy16_g2l(gA0, &As[nxt][ar0][ac0]);
            copy16_g2l(gA1, &As[nxt][ar1][ac1]);
            gA0 += 32; gA1 += 32;
            #pragma unroll
            for (int r = 0; r < 4; ++r) { fb[r] = load_g_f32x4(gB[r]); gB[r] += 32; }
        }

        // ---- fragments + 8 WMMAs on current buffer ----
        v16bf af[4], bfr[2];
        #pragma unroll
        for (int mt = 0; mt < 4; ++mt)
            af[mt] = *(const v16bf*)&As[cur][wm * 64 + mt * 16 + lr][kh];
        #pragma unroll
        for (int nt = 0; nt < 2; ++nt)
            bfr[nt] = *(const v16bf*)&Bs[cur][wn * 32 + nt * 16 + lr][kh];
        #pragma unroll
        for (int mt = 0; mt < 4; ++mt)
            #pragma unroll
            for (int nt = 0; nt < 2; ++nt)
                acc[mt][nt] = __builtin_amdgcn_wmma_f32_16x16x32_bf16(
                    false, af[mt], false, bfr[nt], (short)0, acc[mt][nt],
                    false, false);

        // ---- land next B tile into LDS, wait A DMA, flip ----
        if (has_next) {
            #pragma unroll
            for (int r = 0; r < 4; ++r) {
                bf16x4 hv;
                hv.x = to_bf16(fb[r].x); hv.y = to_bf16(fb[r].y);
                hv.z = to_bf16(fb[r].z); hv.w = to_bf16(fb[r].w);
                *(bf16x4*)&Bs[nxt][brow[r]][bcol[r]] = hv;
            }
            wait_async();
        }
        __syncthreads();
    }
}

// ---------------------------------------------------------------------
// Kernel 1: h = GELU(A1 @ w1^T + b1)    grid = (I/128, TPE/128, E)
// ---------------------------------------------------------------------
__global__ __launch_bounds__(256)
void moe_gemm1_gelu(const bf16_t* __restrict__ A1, const float* __restrict__ w1,
                    const float* __restrict__ b1, bf16_t* __restrict__ Hbuf) {
    const int e  = blockIdx.z;
    const int m0 = blockIdx.y * 128;
    const int n0 = blockIdx.x * 128;

    const bf16_t* A = A1 + ((size_t)e * TPE + m0) * H;
    const float*  B = w1 + (size_t)e * I * H + (size_t)n0 * H;

    v8f acc[4][2] = {};
    int wm, wn, lr, lane;
    wmma_gemm_core<H>(A, B, acc, wm, wn, lr, lane);

    // C layout: VGPR j, lanes 0-15 -> M=j, N=lane; lanes 16-31 -> M=8+j.
    const int mjump = (lane >> 4) << 3;
    #pragma unroll
    for (int mt = 0; mt < 4; ++mt) {
        const int mbase = m0 + wm * 64 + mt * 16 + mjump;
        #pragma unroll
        for (int nt = 0; nt < 2; ++nt) {
            const int ncol = n0 + wn * 32 + nt * 16 + lr;
            const float bias = b1[(size_t)e * I + ncol];
            #pragma unroll
            for (int j = 0; j < 8; ++j) {
                const int m = mbase + j;
                const float x = acc[mt][nt][j] + bias;
                store_g_bf16(Hbuf + ((size_t)e * TPE + m) * I + ncol,
                             to_bf16(gelu_exact(x)));
            }
        }
    }
}

// ---------------------------------------------------------------------
// Kernel 2: o = Hbuf @ w2^T + b2, scattered to Obuf[perm[s]]
// grid = (H/128, TPE/128, E)
// ---------------------------------------------------------------------
__global__ __launch_bounds__(256)
void moe_gemm2_scatter(const bf16_t* __restrict__ Hbuf, const float* __restrict__ w2,
                       const float* __restrict__ b2, const int* __restrict__ perm,
                       float* __restrict__ Obuf) {
    const int e  = blockIdx.z;
    const int m0 = blockIdx.y * 128;
    const int n0 = blockIdx.x * 128;

    const bf16_t* A = Hbuf + ((size_t)e * TPE + m0) * I;
    const float*  B = w2 + (size_t)e * H * I + (size_t)n0 * I;

    v8f acc[4][2] = {};
    int wm, wn, lr, lane;
    wmma_gemm_core<I>(A, B, acc, wm, wn, lr, lane);

    const int mjump = (lane >> 4) << 3;
    #pragma unroll
    for (int mt = 0; mt < 4; ++mt) {
        const int mbase = m0 + wm * 64 + mt * 16 + mjump;
        #pragma unroll
        for (int nt = 0; nt < 2; ++nt) {
            const int ncol = n0 + wn * 32 + nt * 16 + lr;
            const float bias = b2[(size_t)e * H + ncol];
            #pragma unroll
            for (int j = 0; j < 8; ++j) {
                const int m    = mbase + j;
                const int orow = perm[e * TPE + m];      // scatter-restore
                store_g_f32(Obuf + (size_t)orow * H + ncol, acc[mt][nt][j] + bias);
            }
        }
    }
}

// =====================================================================
// Kernel 3: out = LayerNorm(Obuf + hidden) * gamma + beta
// =====================================================================
__global__ __launch_bounds__(256)
void moe_residual_ln(const float* __restrict__ Obuf, const float* __restrict__ hs,
                     const float* __restrict__ gamma, const float* __restrict__ beta,
                     float* __restrict__ out) {
    const int r   = blockIdx.x;
    const int tid = threadIdx.x;
    const f32x4 o = load_g_f32x4(Obuf + (size_t)r * H + tid * 4);
    const f32x4 h = load_g_f32x4(hs   + (size_t)r * H + tid * 4);
    float x0 = o.x + h.x, x1 = o.y + h.y, x2 = o.z + h.z, x3 = o.w + h.w;

    float s  = x0 + x1 + x2 + x3;
    float sq = x0 * x0 + x1 * x1 + x2 * x2 + x3 * x3;
    #pragma unroll
    for (int off = 16; off > 0; off >>= 1) {
        s  += __shfl_xor(s,  off, 32);
        sq += __shfl_xor(sq, off, 32);
    }
    __shared__ float ssum[8], ssq[8];
    __shared__ float mu_s, rs_s;
    if ((tid & 31) == 0) { ssum[tid >> 5] = s; ssq[tid >> 5] = sq; }
    __syncthreads();
    if (tid == 0) {
        float S = 0.f, Q = 0.f;
        #pragma unroll
        for (int i = 0; i < 8; ++i) { S += ssum[i]; Q += ssq[i]; }
        const float mu  = S * (1.0f / H);
        const float var = Q * (1.0f / H) - mu * mu;
        mu_s = mu;
        rs_s = rsqrtf(var + 1e-12f);
    }
    __syncthreads();
    const float mu = mu_s, rs = rs_s;

    const f32x4 g = load_g_f32x4(gamma + tid * 4);
    const f32x4 b = load_g_f32x4(beta  + tid * 4);
    f32x4 y;
    y.x = (x0 - mu) * rs * g.x + b.x;
    y.y = (x1 - mu) * rs * g.y + b.y;
    y.z = (x2 - mu) * rs * g.z + b.z;
    y.w = (x3 - mu) * rs * g.w + b.w;
    *(f32x4 __attribute__((address_space(1)))*)(out + (size_t)r * H + tid * 4) = y;
}

// =====================================================================
// Launch
// =====================================================================
extern "C" void kernel_launch(void* const* d_in, const int* in_sizes, int n_in,
                              void* d_out, int out_size, void* d_ws, size_t ws_size,
                              hipStream_t stream) {
    const float* hidden = (const float*)d_in[0];
    const int*   perm   = (const int*)  d_in[1];
    const float* w1     = (const float*)d_in[2];
    const float* b1     = (const float*)d_in[3];
    const float* w2     = (const float*)d_in[4];
    const float* b2     = (const float*)d_in[5];
    const float* gamma  = (const float*)d_in[6];
    const float* beta   = (const float*)d_in[7];
    float* out = (float*)d_out;

    // workspace layout: A1 bf16 [T,H] | Hbuf bf16 [T,I] | Obuf f32 [T,H]
    char* ws = (char*)d_ws;
    bf16_t* A1   = (bf16_t*)ws;
    bf16_t* Hbuf = (bf16_t*)(ws + (size_t)T * H * sizeof(bf16_t));
    float*  Obuf = (float*) (ws + (size_t)T * H * sizeof(bf16_t)
                                + (size_t)T * I * sizeof(bf16_t));

    moe_gather_cast<<<T, 256, 0, stream>>>(hidden, perm, A1);
    moe_gemm1_gelu<<<dim3(I / 128, TPE / 128, E), 256, 0, stream>>>(A1, w1, b1, Hbuf);
    moe_gemm2_scatter<<<dim3(H / 128, TPE / 128, E), 256, 0, stream>>>(Hbuf, w2, b2, perm, Obuf);
    moe_residual_ln<<<T, 256, 0, stream>>>(Obuf, hidden, gamma, beta, out);
}
